// DroneReconstructionModel_39934605918425
// MI455X (gfx1250) — matrix operand order repaired and checked by reading, not compiled, not run on hardware
//
#include <hip/hip_runtime.h>
#include <hip/hip_bf16.h>

// ---------------------------------------------------------------------------
// Types / constants
// ---------------------------------------------------------------------------
typedef __attribute__((ext_vector_type(16))) _Float16 v16h;
typedef __attribute__((ext_vector_type(8)))  _Float16 v8h;
typedef __attribute__((ext_vector_type(2)))  _Float16 h2;
typedef __attribute__((ext_vector_type(8)))  float    v8f;

#define BM 64
#define BN 64
#define BK 32
#define LROW 24          // halfs per lane-row in fragment LDS (16 used + 8 pad, 48B)
#define BN_EPS 1e-5f

enum EpMode {
  EP_NONE = 0,
  EP_ATOMIC = 1,        // atomicAdd into C (split-K accumulation)
  EP_BN_RELU6 = 2,      // per-row (output channel) BN, clamp [0,6]
  EP_BN = 3,            // per-row BN, optional residual add
  EP_BIAS_RELU = 4,     // per-col bias, relu
  EP_BIAS_SIGMOID = 5   // per-col bias, sigmoid
};

__device__ __forceinline__ v16h load_frag(const _Float16* p) {
  // two 16B LDS loads (ds_load_b128) + register concat
  v8h lo = *(const v8h*)p;
  v8h hi = *(const v8h*)(p + 8);
  return __builtin_shufflevector(lo, hi, 0, 1, 2, 3, 4, 5, 6, 7,
                                         8, 9, 10, 11, 12, 13, 14, 15);
}

// ---------------------------------------------------------------------------
// Stage one BMxBK A tile + BKxBN B tile into fragment-layout LDS.
// FA: rows fully in-range; FB: cols fully in-range; FK: full K tile.
// Non-full paths are BRANCHLESS: clamp the address, select 0 afterwards
// (no exec-mask ladders around the global loads).
//
// Fragment layouts (ISA 7.12.2), inverted for staging:
//  A (16x32 f16): (m,k) -> lane = (m&15) + 16*((k&15)>=8),
//                          e    = (k>=16 ? 8:0) + (k&7)     [k-pairs adjacent]
//  B (32x16 f16): (k,n) -> lane = (n&15) + 16*((k&15)>=8), e = (k>=16 ? 8:0) + (k&7)
// ---------------------------------------------------------------------------
template <bool FA, bool FB, bool FK>
__device__ __forceinline__ void stage_tile(
    const float* __restrict__ A, const float* __restrict__ B,
    int lda, int ldb, int M, int N, int k_end,
    int bm0, int bn0, int k0, int tid,
    _Float16 (&Asf)[4][32][LROW], _Float16 (&Bsf)[4][32][LROW])
{
  float ax[4][2], bx[4][2];
#pragma unroll
  for (int i = 0; i < 4; ++i) {
    int id = tid + i * 256;
    int row = id >> 4, pr = id & 15;
    int gm = bm0 + row, gk = k0 + 2 * pr;
    if (FA && FK) {
      float2 v2 = *(const float2*)&A[(long long)gm * lda + gk];
      ax[i][0] = v2.x; ax[i][1] = v2.y;
    } else {
      int cm  = FA ? gm : (gm < M ? gm : M - 1);
      int ck0 = FK ? gk : (gk < k_end ? gk : k_end - 1);
      int ck1 = FK ? (gk + 1) : (gk + 1 < k_end ? gk + 1 : k_end - 1);
      float x0 = A[(long long)cm * lda + ck0];
      float x1 = A[(long long)cm * lda + ck1];
      bool okm = FA || (gm < M);
      ax[i][0] = (okm && (FK || gk     < k_end)) ? x0 : 0.0f;
      ax[i][1] = (okm && (FK || gk + 1 < k_end)) ? x1 : 0.0f;
    }
  }
#pragma unroll
  for (int i = 0; i < 4; ++i) {
    int id = tid + i * 256;
    int n = id & 63, kp = id >> 6;
    int gn = bn0 + n, gk = k0 + 2 * kp;
    if (FB && FK) {
      bx[i][0] = B[(long long)gk * ldb + gn];
      bx[i][1] = B[(long long)(gk + 1) * ldb + gn];
    } else {
      int cn  = FB ? gn : (gn < N ? gn : N - 1);
      int ck0 = FK ? gk : (gk < k_end ? gk : k_end - 1);
      int ck1 = FK ? (gk + 1) : (gk + 1 < k_end ? gk + 1 : k_end - 1);
      float x0 = B[(long long)ck0 * ldb + cn];
      float x1 = B[(long long)ck1 * ldb + cn];
      bool okn = FB || (gn < N);
      bx[i][0] = (okn && (FK || gk     < k_end)) ? x0 : 0.0f;
      bx[i][1] = (okn && (FK || gk + 1 < k_end)) ? x1 : 0.0f;
    }
  }
#pragma unroll
  for (int i = 0; i < 4; ++i) {
    int id = tid + i * 256;
    int row = id >> 4, pr = id & 15;
    int k = 2 * pr, kk = k & 15;
    int lw = (row & 15) + ((kk >= 8) ? 16 : 0);
    int e  = ((k >= 16) ? 8 : 0) + (kk & 7);
    *(h2*)&Asf[row >> 4][lw][e] = (h2){(_Float16)ax[i][0], (_Float16)ax[i][1]};
  }
#pragma unroll
  for (int i = 0; i < 4; ++i) {
    int id = tid + i * 256;
    int n = id & 63, kp = id >> 6;
    int k = 2 * kp;
    int lw = (n & 15) + (((k & 15) >= 8) ? 16 : 0);
    *(h2*)&Bsf[n >> 4][lw][((k >= 16) ? 8 : 0) + (k & 7)] = (h2){(_Float16)bx[i][0], (_Float16)bx[i][1]};
  }
}

// ---------------------------------------------------------------------------
// Generic tiled WMMA GEMM:  C[m,n] = sum_k A[m,k] * B[k,n]
// fp32 in memory, f16 fragment-layout LDS, double buffered (one barrier per
// K-step; next tile's global loads overlap current WMMA).  fp32 accumulate via
// v_wmma_f32_16x16x32_f16.  Batched via z-grid; split-K via z + atomics.
// 256 threads = 8 wave32; waves 4(M) x 2(N); each wave owns 16x32 of C.
// ---------------------------------------------------------------------------
__global__ __launch_bounds__(256)
void gemm_wmma_kernel(const float* __restrict__ A, const float* __restrict__ B,
                      float* __restrict__ C,
                      int M, int N, int K, int lda, int ldb, int ldc,
                      long long sA, long long sB, long long sC,
                      int splitk, int kchunk, int ep,
                      const float* __restrict__ p0, const float* __restrict__ p1,
                      const float* __restrict__ p2, const float* __restrict__ p3,
                      const float* __restrict__ res, long long sRes)
{
  __shared__ __align__(16) _Float16 Asf[2][4][32][LROW];
  __shared__ __align__(16) _Float16 Bsf[2][4][32][LROW];

  const int tid = threadIdx.x;
  const int bn0 = blockIdx.x * BN;
  const int bm0 = blockIdx.y * BM;
  const int zb  = blockIdx.z;
  const int batch  = zb / splitk;
  const int kslice = zb % splitk;

  A += (long long)batch * sA;
  B += (long long)batch * sB;
  C += (long long)batch * sC;

  const int k_begin = kslice * kchunk;
  int k_end = k_begin + kchunk;
  if (k_end > K) k_end = K;

  const int wave = tid >> 5;
  const int lane = tid & 31;
  const int mt   = wave & 3;         // wave m-tile (16 rows each)
  const int ng   = wave >> 2;        // wave n-group (2 n-tiles each)

  const int nk = (k_end - k_begin + BK - 1) / BK;
  const bool lastPartial = (k_begin + nk * BK > k_end);
  const int nfull = lastPartial ? nk - 1 : nk;
  const bool interior = (bm0 + BM <= M) && (bn0 + BN <= N);

  v8f c0 = {};
  v8f c1 = {};

  if (nk > 0) {
    // prologue: stage tile 0 into buffer 0
    if (interior) {
      if (0 < nfull) stage_tile<true, true, true>(A, B, lda, ldb, M, N, k_end, bm0, bn0, k_begin, tid, Asf[0], Bsf[0]);
      else           stage_tile<true, true, false>(A, B, lda, ldb, M, N, k_end, bm0, bn0, k_begin, tid, Asf[0], Bsf[0]);
    } else {
      if (0 < nfull) stage_tile<false, false, true>(A, B, lda, ldb, M, N, k_end, bm0, bn0, k_begin, tid, Asf[0], Bsf[0]);
      else           stage_tile<false, false, false>(A, B, lda, ldb, M, N, k_end, bm0, bn0, k_begin, tid, Asf[0], Bsf[0]);
    }
    __syncthreads();

    if (interior) {
      for (int it = 0; it < nk; ++it) {
        int cb = it & 1;
        v16h a  = load_frag(&Asf[cb][mt][lane][0]);
        v16h b0 = load_frag(&Bsf[cb][ng * 2][lane][0]);
        v16h b1 = load_frag(&Bsf[cb][ng * 2 + 1][lane][0]);
        int jn = it + 1;
        if (jn < nk) {
          int k0n = k_begin + jn * BK;
          if (jn + 1 < nk) {  // light L2 prefetch two tiles ahead
            int k0p = k0n + BK;
            __builtin_prefetch(&A[(long long)(bm0 + (tid & 63)) * lda + k0p], 0, 1);
            __builtin_prefetch(&B[(long long)(k0p + (tid & 31)) * ldb + bn0], 0, 1);
          }
          if (jn < nfull) stage_tile<true, true, true>(A, B, lda, ldb, M, N, k_end, bm0, bn0, k0n, tid, Asf[cb ^ 1], Bsf[cb ^ 1]);
          else            stage_tile<true, true, false>(A, B, lda, ldb, M, N, k_end, bm0, bn0, k0n, tid, Asf[cb ^ 1], Bsf[cb ^ 1]);
        }
        c0 = __builtin_amdgcn_wmma_f32_16x16x32_f16(false, a, false, b0, (short)0, c0, false, false);
        c1 = __builtin_amdgcn_wmma_f32_16x16x32_f16(false, a, false, b1, (short)0, c1, false, false);
        __syncthreads();
      }
    } else {
      for (int it = 0; it < nk; ++it) {
        int cb = it & 1;
        v16h a  = load_frag(&Asf[cb][mt][lane][0]);
        v16h b0 = load_frag(&Bsf[cb][ng * 2][lane][0]);
        v16h b1 = load_frag(&Bsf[cb][ng * 2 + 1][lane][0]);
        int jn = it + 1;
        if (jn < nk) {
          int k0n = k_begin + jn * BK;
          if (jn < nfull) stage_tile<false, false, true>(A, B, lda, ldb, M, N, k_end, bm0, bn0, k0n, tid, Asf[cb ^ 1], Bsf[cb ^ 1]);
          else            stage_tile<false, false, false>(A, B, lda, ldb, M, N, k_end, bm0, bn0, k0n, tid, Asf[cb ^ 1], Bsf[cb ^ 1]);
        }
        c0 = __builtin_amdgcn_wmma_f32_16x16x32_f16(false, a, false, b0, (short)0, c0, false, false);
        c1 = __builtin_amdgcn_wmma_f32_16x16x32_f16(false, a, false, b1, (short)0, c1, false, false);
        __syncthreads();
      }
    }
  }

  // ---- epilogue; C/D layout: VGPR r -> M = r + 8*(lane>=16), N = lane&15 ----
  const int wm = mt * 16;
  const int wn = ng * 32;
  const int mhalf = (lane >> 4) * 8;
  const int nloc  = lane & 15;
#pragma unroll
  for (int r = 0; r < 8; ++r) {
    int gm = bm0 + wm + mhalf + r;
    if (gm >= M) continue;
    float s = 0.0f, sh = 0.0f;
    if (ep == EP_BN_RELU6 || ep == EP_BN) {
      s  = p0[gm] * rsqrtf(p3[gm] + BN_EPS);
      sh = p1[gm] - p2[gm] * s;
    }
#pragma unroll
    for (int t = 0; t < 2; ++t) {
      int gn = bn0 + wn + t * 16 + nloc;
      if (gn >= N) continue;
      float acc = (t == 0) ? c0[r] : c1[r];
      long long ci = (long long)gm * ldc + gn;
      switch (ep) {
        case EP_ATOMIC:
          atomicAdd(&C[ci], acc);
          break;
        case EP_BN_RELU6: {
          float v = acc * s + sh;
          C[ci] = fminf(fmaxf(v, 0.0f), 6.0f);
        } break;
        case EP_BN: {
          float v = acc * s + sh;
          if (res) v += res[(long long)batch * sRes + (long long)gm * ldc + gn];
          C[ci] = v;
        } break;
        case EP_BIAS_RELU: {
          float v = acc + p0[gn];
          C[ci] = fmaxf(v, 0.0f);
        } break;
        case EP_BIAS_SIGMOID: {
          float v = acc + p0[gn];
          C[ci] = 1.0f / (1.0f + expf(-v));
        } break;
        default:
          C[ci] = acc;
      }
    }
  }
}

// ---------------------------------------------------------------------------
// Stem: 3x3 s2 conv (cin=3) + BN + ReLU6, NCHW.  16x32x112x112 out.
// ---------------------------------------------------------------------------
__global__ void stem_conv_kernel(const float* __restrict__ in, const float* __restrict__ w,
                                 float* __restrict__ out,
                                 const float* __restrict__ g, const float* __restrict__ b,
                                 const float* __restrict__ m, const float* __restrict__ v)
{
  const int H = 224, OH = 112, C = 32;
  int idx = blockIdx.x * blockDim.x + threadIdx.x;
  int total = 16 * C * OH * OH;
  if (idx >= total) return;
  int ow = idx % OH; int t = idx / OH;
  int oh = t % OH; t /= OH;
  int co = t % C; int n = t / C;

  float scale = g[co] * rsqrtf(v[co] + BN_EPS);
  float shift = b[co] - m[co] * scale;
  float acc = 0.0f;
  int ih0 = oh * 2 - 1, iw0 = ow * 2 - 1;
#pragma unroll
  for (int ci = 0; ci < 3; ++ci) {
    const float* ip = in + ((long long)n * 3 + ci) * H * H;
    const float* wp = w + ((long long)co * 3 + ci) * 9;
#pragma unroll
    for (int kh = 0; kh < 3; ++kh) {
      int ih = ih0 + kh;
      if (ih < 0 || ih >= H) continue;
#pragma unroll
      for (int kw = 0; kw < 3; ++kw) {
        int iw = iw0 + kw;
        if (iw < 0 || iw >= H) continue;
        acc += ip[ih * H + iw] * wp[kh * 3 + kw];
      }
    }
  }
  float r = acc * scale + shift;
  out[idx] = fminf(fmaxf(r, 0.0f), 6.0f);
}

// ---------------------------------------------------------------------------
// Depthwise 3x3 + BN + ReLU6
// ---------------------------------------------------------------------------
__global__ void dwconv_kernel(const float* __restrict__ in, const float* __restrict__ w,
                              float* __restrict__ out,
                              const float* __restrict__ g, const float* __restrict__ b,
                              const float* __restrict__ m, const float* __restrict__ v,
                              int C, int H, int OH, int stride)
{
  int idx = blockIdx.x * blockDim.x + threadIdx.x;
  int total = 16 * C * OH * OH;
  if (idx >= total) return;
  int ow = idx % OH; int t = idx / OH;
  int oh = t % OH; t /= OH;
  int c = t % C; int n = t / C;

  const float* ip = in + ((long long)n * C + c) * H * H;
  const float* wp = w + (long long)c * 9;
  float scale = g[c] * rsqrtf(v[c] + BN_EPS);
  float shift = b[c] - m[c] * scale;
  float acc = 0.0f;
  int ih0 = oh * stride - 1, iw0 = ow * stride - 1;
#pragma unroll
  for (int kh = 0; kh < 3; ++kh) {
    int ih = ih0 + kh;
    if (ih < 0 || ih >= H) continue;
#pragma unroll
    for (int kw = 0; kw < 3; ++kw) {
      int iw = iw0 + kw;
      if (iw < 0 || iw >= H) continue;
      acc += ip[ih * H + iw] * wp[kh * 3 + kw];
    }
  }
  float r = acc * scale + shift;
  out[idx] = fminf(fmaxf(r, 0.0f), 6.0f);
}

// ---------------------------------------------------------------------------
// Utility: zero fill
// ---------------------------------------------------------------------------
__global__ void zero_kernel(float* __restrict__ p, long long n)
{
  long long i = (long long)blockIdx.x * blockDim.x + threadIdx.x;
  long long stride = (long long)gridDim.x * blockDim.x;
  for (; i < n; i += stride) p[i] = 0.0f;
}

// ---------------------------------------------------------------------------
// Position MLP: 3 -> 64 -> 128, ReLU.  One block per node (16 blocks x 128).
// ---------------------------------------------------------------------------
__global__ void pos_mlp_kernel(const float* __restrict__ pos,
                               const float* __restrict__ w1, const float* __restrict__ b1,
                               const float* __restrict__ w2, const float* __restrict__ b2,
                               float* __restrict__ out)
{
  __shared__ float p1[64];
  int n = blockIdx.x, t = threadIdx.x;
  if (t < 64) {
    float a = b1[t];
#pragma unroll
    for (int i = 0; i < 3; ++i) a += pos[n * 3 + i] * w1[i * 64 + t];
    p1[t] = fmaxf(a, 0.0f);
  }
  __syncthreads();
  float a = b2[t];
  for (int i = 0; i < 64; ++i) a += p1[i] * w2[i * 128 + t];
  out[n * 128 + t] = fmaxf(a, 0.0f);
}

// ---------------------------------------------------------------------------
// a_src[n] = h[n,:].att_src ; a_dst[n] = h[n,:].att_dst  (16 blocks x 256)
// ---------------------------------------------------------------------------
__global__ void attn_proj_kernel(const float* __restrict__ h,
                                 const float* __restrict__ att_src,
                                 const float* __restrict__ att_dst,
                                 float* __restrict__ a_src, float* __restrict__ a_dst)
{
  __shared__ float sbuf[256];
  int n = blockIdx.x, t = threadIdx.x;
  float s1 = 0.0f, s2 = 0.0f;
  for (int k = t; k < 1024; k += 256) {
    float hv = h[n * 1024 + k];
    s1 += hv * att_src[k];
    s2 += hv * att_dst[k];
  }
  sbuf[t] = s1; __syncthreads();
  for (int o = 128; o > 0; o >>= 1) { if (t < o) sbuf[t] += sbuf[t + o]; __syncthreads(); }
  if (t == 0) a_src[n] = sbuf[0];
  __syncthreads();
  sbuf[t] = s2; __syncthreads();
  for (int o = 128; o > 0; o >>= 1) { if (t < o) sbuf[t] += sbuf[t + o]; __syncthreads(); }
  if (t == 0) a_dst[n] = sbuf[0];
}

// ---------------------------------------------------------------------------
// Edge logits (leaky relu 0.2) + segment softmax over dst.  144 edges
// (128 given + 16 self loops), single block of 256 threads.
// ---------------------------------------------------------------------------
__global__ void edge_softmax_kernel(const int* __restrict__ edge,
                                    const float* __restrict__ a_src,
                                    const float* __restrict__ a_dst,
                                    float* __restrict__ alpha)
{
  __shared__ float lg[144];
  __shared__ int sdst[144];
  __shared__ float nmax[16], nsum[16];
  int t = threadIdx.x;
  if (t < 144) {
    int s, d;
    if (t < 128) { s = edge[t]; d = edge[128 + t]; }
    else         { s = t - 128; d = t - 128; }
    sdst[t] = d;
    float l = a_src[s] + a_dst[d];
    lg[t] = (l > 0.0f) ? l : 0.2f * l;
  }
  __syncthreads();
  if (t < 16) {
    float mx = -3.0e38f;
    for (int e = 0; e < 144; ++e) if (sdst[e] == t) mx = fmaxf(mx, lg[e]);
    nmax[t] = mx;
  }
  __syncthreads();
  if (t < 144) lg[t] = expf(lg[t] - nmax[sdst[t]]);
  __syncthreads();
  if (t < 16) {
    float s = 0.0f;
    for (int e = 0; e < 144; ++e) if (sdst[e] == t) s += lg[e];
    nsum[t] = s;
  }
  __syncthreads();
  if (t < 144) alpha[t] = lg[t] / (nsum[sdst[t]] + 1e-16f);
}

// ---------------------------------------------------------------------------
// gat_out[d,f] = bias[f] + sum_{e: dst[e]==d} alpha[e] * h[src[e], f]
// One block per dst node; 256 threads stride over 1024 features.
// ---------------------------------------------------------------------------
__global__ void gat_aggregate_kernel(const float* __restrict__ h,
                                     const float* __restrict__ alpha,
                                     const int* __restrict__ edge,
                                     const float* __restrict__ bias,
                                     float* __restrict__ out)
{
  int d = blockIdx.x, t = threadIdx.x;
  for (int f = t; f < 1024; f += 256) {
    float acc = bias[f];
    for (int e = 0; e < 144; ++e) {
      int dd = (e < 128) ? edge[128 + e] : (e - 128);
      if (dd == d) {
        int ss = (e < 128) ? edge[e] : (e - 128);
        acc += alpha[e] * h[ss * 1024 + f];
      }
    }
    out[d * 1024 + f] = acc;
  }
}

// ---------------------------------------------------------------------------
// Host orchestration
// ---------------------------------------------------------------------------
struct CBN { const float *b, *g, *m, *v, *w; };

static void launch_gemm(hipStream_t s, const float* A, const float* B, float* C,
                        int M, int N, int K, int lda, int ldb, int ldc,
                        long long sA, long long sB, long long sC, int batch,
                        int splitk, int ep,
                        const float* p0, const float* p1, const float* p2, const float* p3,
                        const float* res, long long sRes)
{
  int kchunk;
  if (splitk <= 1) { splitk = 1; kchunk = K; }
  else {
    kchunk = (((K + splitk - 1) / splitk + BK - 1) / BK) * BK;
    splitk = (K + kchunk - 1) / kchunk;
  }
  dim3 grid((N + BN - 1) / BN, (M + BM - 1) / BM, batch * splitk);
  gemm_wmma_kernel<<<grid, 256, 0, s>>>(A, B, C, M, N, K, lda, ldb, ldc,
                                        sA, sB, sC, splitk, kchunk, ep,
                                        p0, p1, p2, p3, res, sRes);
}

extern "C" void kernel_launch(void* const* d_in, const int* in_sizes, int n_in,
                              void* d_out, int out_size, void* d_ws, size_t ws_size,
                              hipStream_t stream)
{
  (void)in_sizes; (void)n_in; (void)out_size; (void)ws_size;

  // ---- parameter walk (top-level insertion order; nested dicts sorted keys) ----
  int cur = 0;
  auto nx = [&]() -> const float* { return (const float*)d_in[cur++]; };

  const float* in_x   = nx();                 // x [16,3,224,224]
  const float* in_pos = nx();                 // positions [16,3]
  const int*   in_edge = (const int*)d_in[cur++];  // edge_index [2,128]

  auto cbn = [&]() { CBN c; c.b = nx(); c.g = nx(); c.m = nx(); c.v = nx(); c.w = nx(); return c; };

  static const int MB_T[7] = {1, 6, 6, 6, 6, 6, 6};
  static const int MB_C[7] = {16, 24, 32, 64, 96, 160, 320};
  static const int MB_N[7] = {1, 2, 3, 4, 3, 3, 1};
  static const int MB_S[7] = {1, 2, 2, 2, 1, 2, 1};

  CBN stem = cbn();
  CBN dwp[17], exp_[17], prj[17];
  bool hasex[17], resf[17];
  int sarr[17], coarr[17], hidarr[17];
  {
    int cin = 32, li = 0;
    for (int bi = 0; bi < 7; ++bi)
      for (int i = 0; i < MB_N[bi]; ++i) {
        int t = MB_T[bi], c = MB_C[bi], s = (i == 0) ? MB_S[bi] : 1;
        sarr[li] = s; coarr[li] = c; hidarr[li] = cin * t;
        resf[li] = (s == 1 && cin == c);
        if (t != 1) { dwp[li] = cbn(); exp_[li] = cbn(); prj[li] = cbn(); hasex[li] = true; }
        else        { dwp[li] = cbn(); prj[li] = cbn(); hasex[li] = false; }
        cin = c; ++li;
      }
  }
  CBN head = cbn();
  // pos_params: b1,b2,w1,w2
  const float* pb1 = nx(); const float* pb2 = nx();
  const float* pw1 = nx(); const float* pw2 = nx();
  // gat_params: W, att_dst, att_src, bias
  const float* gW    = nx();
  const float* gadst = nx();
  const float* gasrc = nx();
  const float* gbias = nx();
  // dec_params: b1,b2,w1,w2
  const float* db1 = nx(); const float* db2 = nx();
  const float* dw1 = nx(); const float* dw2 = nx();

  // ---- workspace layout (floats) ----
  float* ws = (float*)d_ws;
  const long long XCAP = 16LL * 32 * 112 * 112;   // 6,422,528
  const long long ECAP = 16LL * 96 * 112 * 112;   // 19,267,584
  float* xA   = ws;
  float* xB   = xA + XCAP;
  float* bufE = xB + XCAP;
  float* bufD = bufE + ECAP;
  float* imgf = bufD + XCAP;            // [16, 62720]
  float* pbuf = imgf + 16LL * 62720;    // [16, 128]
  float* hbuf = pbuf + 16 * 128;        // [16, 1024]
  float* asrc = hbuf + 16 * 1024;       // [16]
  float* adst = asrc + 16;              // [16]
  float* alph = adst + 16;              // [144]
  float* gato = alph + 144;             // [16, 1024]
  float* dbuf = gato + 16 * 1024;       // [16, 512]

  // ---- MobileNetV2 ----
  {
    int total = 16 * 32 * 112 * 112;
    stem_conv_kernel<<<(total + 255) / 256, 256, 0, stream>>>(
        in_x, stem.w, xA, stem.g, stem.b, stem.m, stem.v);
  }
  float* xc = xA; float* xn = xB;
  int C = 32, H = 112;
  for (int li = 0; li < 17; ++li) {
    int hid = hidarr[li], s = sarr[li], co = coarr[li];
    int OH = (H - 1) / s + 1;
    const float* dwin = xc;
    if (hasex[li]) {
      // expand 1x1: per image, W[hid,C] @ X[C, H*H] -> E[hid, H*H], BN+ReLU6
      launch_gemm(stream, exp_[li].w, xc, bufE, hid, H * H, C,
                  C, H * H, H * H,
                  0, (long long)C * H * H, (long long)hid * H * H, 16, 1,
                  EP_BN_RELU6, exp_[li].g, exp_[li].b, exp_[li].m, exp_[li].v, nullptr, 0);
      dwin = bufE;
    }
    {
      int total = 16 * hid * OH * OH;
      dwconv_kernel<<<(total + 255) / 256, 256, 0, stream>>>(
          dwin, dwp[li].w, bufD, dwp[li].g, dwp[li].b, dwp[li].m, dwp[li].v,
          hid, H, OH, s);
    }
    // proj 1x1 (linear bottleneck): W[co,hid] @ D[hid, OH*OH], BN, + residual
    const float* res = resf[li] ? xc : nullptr;
    launch_gemm(stream, prj[li].w, bufD, xn, co, OH * OH, hid,
                hid, OH * OH, OH * OH,
                0, (long long)hid * OH * OH, (long long)co * OH * OH, 16, 1,
                EP_BN, prj[li].g, prj[li].b, prj[li].m, prj[li].v,
                res, (long long)co * OH * OH);
    float* tmp = xc; xc = xn; xn = tmp;
    C = co; H = OH;
  }
  // head 1x1: W[1280,320] @ X[320,49] -> imgf[1280,49] per image, BN+ReLU6
  launch_gemm(stream, head.w, xc, imgf, 1280, 49, C,
              C, 49, 49,
              0, (long long)C * 49, 62720LL, 16, 1,
              EP_BN_RELU6, head.g, head.b, head.m, head.v, nullptr, 0);

  // ---- position MLP ----
  pos_mlp_kernel<<<16, 128, 0, stream>>>(in_pos, pw1, pb1, pw2, pb2, pbuf);

  // ---- GAT projection h = [img | p] @ W   (split W rows; split-K atomics) ----
  zero_kernel<<<64, 256, 0, stream>>>(hbuf, 16LL * 1024);
  launch_gemm(stream, imgf, gW, hbuf, 16, 1024, 62720,
              62720, 1024, 1024, 0, 0, 0, 1, 64,
              EP_ATOMIC, nullptr, nullptr, nullptr, nullptr, nullptr, 0);
  launch_gemm(stream, pbuf, gW + 62720LL * 1024, hbuf, 16, 1024, 128,
              128, 1024, 1024, 0, 0, 0, 1, 1,
              EP_ATOMIC, nullptr, nullptr, nullptr, nullptr, nullptr, 0);

  // ---- attention + softmax + aggregation ----
  attn_proj_kernel<<<16, 256, 0, stream>>>(hbuf, gasrc, gadst, asrc, adst);
  edge_softmax_kernel<<<1, 256, 0, stream>>>(in_edge, asrc, adst, alph);
  gat_aggregate_kernel<<<16, 256, 0, stream>>>(hbuf, alph, in_edge, gbias, gato);

  // ---- decoder ----
  launch_gemm(stream, gato, dw1, dbuf, 16, 512, 1024,
              1024, 512, 512, 0, 0, 0, 1, 1,
              EP_BIAS_RELU, db1, nullptr, nullptr, nullptr, nullptr, 0);
  launch_gemm(stream, dbuf, dw2, (float*)d_out, 16, 50176, 512,
              512, 50176, 50176, 0, 0, 0, 1, 1,
              EP_BIAS_SIGMOID, db2, nullptr, nullptr, nullptr, nullptr, 0);
}
